// QuantizedQwen2MLP_71725953843877
// MI455X (gfx1250) — 8-bit matrix operand order repaired and checked by reading, compiled, not run on hardware
//
#include <hip/hip_runtime.h>
#include <cstdint>
#include <cstddef>

typedef int v8i __attribute__((ext_vector_type(8)));

#define QMAXF 127.0f

__device__ __forceinline__ v8i load_a_frag(const int8_t* __restrict__ p) {
  int2 a0 = *(const int2*)(p);
  int2 a1 = *(const int2*)(p + 16);
  int2 a2 = *(const int2*)(p + 32);
  int2 a3 = *(const int2*)(p + 48);
  v8i A = {a0.x, a0.y, a1.x, a1.y, a2.x, a2.y, a3.x, a3.y};
  return A;
}

__device__ __forceinline__ v8i load_b_frag(const int8_t* __restrict__ p) {
  int2 b0 = *(const int2*)(p);
  int2 b1 = *(const int2*)(p + 16);
  int2 b2 = *(const int2*)(p + 32);
  int2 b3 = *(const int2*)(p + 48);
  v8i B = {b0.x, b0.y, b1.x, b1.y, b2.x, b2.y, b3.x, b3.y};
  return B;
}

__global__ __launch_bounds__(256) void pack_w_int8(const int* __restrict__ w,
                                                   int8_t* __restrict__ q,
                                                   int n4) {
  int stride = gridDim.x * blockDim.x;
  for (int i = blockIdx.x * blockDim.x + threadIdx.x; i < n4; i += stride) {
    int4 v = ((const int4*)w)[i];
    int packed = (v.x & 255) | ((v.y & 255) << 8) | ((v.z & 255) << 16)
               | ((v.w & 255) << 24);
    ((int*)q)[i] = packed;
  }
}

__global__ __launch_bounds__(256) void quant_x_kernel(const float* __restrict__ x,
                                                      int8_t* __restrict__ qx,
                                                      float* __restrict__ xs,
                                                      int H) {
  const int t = blockIdx.x;
  const float* xr = x + (size_t)t * H;
  __shared__ float red[256];
  float mx = 0.0f;
  for (int h = threadIdx.x; h < H; h += 256) mx = fmaxf(mx, fabsf(xr[h]));
  red[threadIdx.x] = mx;
  __syncthreads();
  for (int s = 128; s > 0; s >>= 1) {
    if (threadIdx.x < s) red[threadIdx.x] = fmaxf(red[threadIdx.x], red[threadIdx.x + s]);
    __syncthreads();
  }
  const float scale = fmaxf(red[0] / QMAXF, 1e-8f);
  if (threadIdx.x == 0) xs[t] = scale;
  const float inv = 1.0f / scale;
  int* qrow = (int*)(qx + (size_t)t * H);
  const int n4 = H >> 2;
  for (int i = threadIdx.x; i < n4; i += 256) {
    float4 v = ((const float4*)xr)[i];
    int a = (int)fminf(fmaxf(rintf(v.x * inv), -QMAXF), QMAXF);
    int b = (int)fminf(fmaxf(rintf(v.y * inv), -QMAXF), QMAXF);
    int c = (int)fminf(fmaxf(rintf(v.z * inv), -QMAXF), QMAXF);
    int d = (int)fminf(fmaxf(rintf(v.w * inv), -QMAXF), QMAXF);
    qrow[i] = (a & 255) | ((b & 255) << 8) | ((c & 255) << 16) | ((d & 255) << 24);
  }
}

__global__ __launch_bounds__(256) void gateup_wmma_kernel(
    const int8_t* __restrict__ qx, const int8_t* __restrict__ qwg,
    const int8_t* __restrict__ qwu, const float* __restrict__ xs,
    const float* __restrict__ s_gate, const float* __restrict__ s_up,
    const float* __restrict__ dscale_p, int8_t* __restrict__ qh,
    int H, int I) {
  const int lane = threadIdx.x & 31;
  const int wave = threadIdx.x >> 5;
  const int n2 = blockIdx.x * 8 + wave;
  const int m4 = blockIdx.y;
  const int t0 = m4 << 6;
  const int i0 = n2 << 5;
  const int half = lane >> 4;
  const int l16 = lane & 15;

  const int8_t* arow = qx + (size_t)(t0 + l16) * H + (half << 3);
  const int8_t* gp = qwg + (size_t)(i0 + l16) * H + (half << 3);
  const int8_t* up = qwu + (size_t)(i0 + l16) * H + (half << 3);
  const size_t rstep = (size_t)H << 4;

  v8i gacc[8];
  v8i uacc[8];
#pragma unroll
  for (int i = 0; i < 8; ++i) {
    gacc[i] = (v8i){0, 0, 0, 0, 0, 0, 0, 0};
    uacc[i] = (v8i){0, 0, 0, 0, 0, 0, 0, 0};
  }

#pragma unroll 1
  for (int kb = 0; kb < H; kb += 64) {
    v8i A[4];
#pragma unroll
    for (int mi = 0; mi < 4; ++mi) A[mi] = load_a_frag(arow + (size_t)mi * rstep + kb);
    v8i Bg0 = load_b_frag(gp + kb);
    v8i Bg1 = load_b_frag(gp + rstep + kb);
    v8i Bu0 = load_b_frag(up + kb);
    v8i Bu1 = load_b_frag(up + rstep + kb);

#pragma unroll
    for (int mi = 0; mi < 4; ++mi) {
      gacc[mi * 2 + 0] = __builtin_amdgcn_wmma_i32_16x16x64_iu8(true, A[mi], true, Bg0, gacc[mi * 2 + 0], false, false);
      gacc[mi * 2 + 1] = __builtin_amdgcn_wmma_i32_16x16x64_iu8(true, A[mi], true, Bg1, gacc[mi * 2 + 1], false, false);
      uacc[mi * 2 + 0] = __builtin_amdgcn_wmma_i32_16x16x64_iu8(true, A[mi], true, Bu0, uacc[mi * 2 + 0], false, false);
      uacc[mi * 2 + 1] = __builtin_amdgcn_wmma_i32_16x16x64_iu8(true, A[mi], true, Bu1, uacc[mi * 2 + 1], false, false);
    }
  }

  const float invd = 1.0f / dscale_p[0];
#pragma unroll
  for (int ni = 0; ni < 2; ++ni) {
    const int icol = i0 + (ni << 4) + l16;
    const float sgf = s_gate[icol];
    const float suf = s_up[icol];
    int8_t* qcol = qh + icol;
#pragma unroll
    for (int mi = 0; mi < 4; ++mi) {
#pragma unroll
      for (int r = 0; r < 8; ++r) {
        const int t = t0 + (mi << 4) + r + (half << 3);
        const float xscale = xs[t];
        const float g = (float)gacc[mi * 2 + ni][r] * xscale * sgf;
        const float u = (float)uacc[mi * 2 + ni][r] * xscale * suf;
        const float hval = (g / (1.0f + __expf(-g))) * u;
        const float q = fminf(fmaxf(rintf(hval * invd), -QMAXF), QMAXF);
        qcol[(size_t)t * I] = (int8_t)(int)q;
      }
    }
  }
}

__global__ __launch_bounds__(256) void down_wmma_kernel(
    const int8_t* __restrict__ qh, const int8_t* __restrict__ qwd,
    const float* __restrict__ s_down, const float* __restrict__ dscale_p,
    float* __restrict__ out, int H, int I) {
  const int lane = threadIdx.x & 31;
  const int wave = threadIdx.x >> 5;
  const int n4 = blockIdx.x * 8 + wave;
  const int m4 = blockIdx.y;
  const int t0 = m4 << 6;
  const int h0 = n4 << 6;
  const int half = lane >> 4;
  const int l16 = lane & 15;

  const int8_t* arow = qh + (size_t)(t0 + l16) * I + (half << 3);
  const int8_t* brow = qwd + (size_t)(h0 + l16) * I + (half << 3);
  const size_t rstep = (size_t)I << 4;

  v8i acc[16];
#pragma unroll
  for (int i = 0; i < 16; ++i) acc[i] = (v8i){0, 0, 0, 0, 0, 0, 0, 0};

#pragma unroll 1
  for (int kb = 0; kb < I; kb += 64) {
    v8i A[4];
#pragma unroll
    for (int mi = 0; mi < 4; ++mi) A[mi] = load_a_frag(arow + (size_t)mi * rstep + kb);
    v8i B[4];
#pragma unroll
    for (int ni = 0; ni < 4; ++ni) B[ni] = load_b_frag(brow + (size_t)ni * rstep + kb);
#pragma unroll
    for (int mi = 0; mi < 4; ++mi)
#pragma unroll
      for (int ni = 0; ni < 4; ++ni)
        acc[mi * 4 + ni] = __builtin_amdgcn_wmma_i32_16x16x64_iu8(true, A[mi], true, B[ni], acc[mi * 4 + ni], false, false);
  }

  const float dscale = dscale_p[0];
#pragma unroll
  for (int ni = 0; ni < 4; ++ni) {
    const int hcol = h0 + (ni << 4) + l16;
    const float sc = dscale * s_down[hcol];
    float* ocol = out + hcol;
#pragma unroll
    for (int mi = 0; mi < 4; ++mi) {
#pragma unroll
      for (int r = 0; r < 8; ++r) {
        const int t = t0 + (mi << 4) + r + (half << 3);
        ocol[(size_t)t * H] = (float)acc[mi * 4 + ni][r] * sc;
      }
    }
  }
}

extern "C" void kernel_launch(void* const* d_in, const int* in_sizes, int n_in,
                              void* d_out, int out_size, void* d_ws, size_t ws_size,
                              hipStream_t stream) {
  const float* x        = (const float*)d_in[0];
  const int*   w_gate   = (const int*)d_in[1];
  const int*   w_up     = (const int*)d_in[2];
  const int*   w_down   = (const int*)d_in[3];
  const float* s_gate   = (const float*)d_in[4];
  const float* s_up     = (const float*)d_in[5];
  const float* s_down   = (const float*)d_in[6];
  const float* dscale_p = (const float*)d_in[7];

  const int I = in_sizes[4];
  const int H = in_sizes[6];
  const int T = in_sizes[0] / H;

  uint8_t* ws = (uint8_t*)d_ws;
  size_t off = 0;
  int8_t* qx  = (int8_t*)(ws + off); off += (size_t)T * H;
  int8_t* qwg = (int8_t*)(ws + off); off += (size_t)I * H;
  int8_t* qwu = (int8_t*)(ws + off); off += (size_t)I * H;
  int8_t* qwd = (int8_t*)(ws + off); off += (size_t)H * I;
  int8_t* qh  = (int8_t*)(ws + off); off += (size_t)T * I;
  float*  xs  = (float*)(ws + off);  off += (size_t)T * 4;

  {
    const int n4w = (int)(((size_t)I * H) >> 2);
    pack_w_int8<<<2048, 256, 0, stream>>>(w_gate, qwg, n4w);
    pack_w_int8<<<2048, 256, 0, stream>>>(w_up,   qwu, n4w);
    pack_w_int8<<<2048, 256, 0, stream>>>(w_down, qwd, n4w);
  }

  quant_x_kernel<<<T, 256, 0, stream>>>(x, qx, xs, H);

  {
    dim3 grid(I / 32 / 8, T / 64);
    gateup_wmma_kernel<<<grid, 256, 0, stream>>>(qx, qwg, qwu, xs, s_gate,
                                                 s_up, dscale_p, qh, H, I);
  }

  {
    dim3 grid(H / 64 / 8, T / 64);
    down_wmma_kernel<<<grid, 256, 0, stream>>>(qh, qwd, s_down, dscale_p,
                                               (float*)d_out, H, I);
  }
}
